// InteractionModule_60627758350972
// MI455X (gfx1250) — compile-verified, matrix operand order checked by reading, not run on hardware
//
#include <hip/hip_runtime.h>
#include <math.h>

typedef _Float16 f16;
typedef __attribute__((ext_vector_type(8)))  _Float16 v8h;
typedef __attribute__((ext_vector_type(16))) _Float16 v16h;
typedef __attribute__((ext_vector_type(8)))  float    v8f;

// ---------------------------------------------------------------------------
// Problem constants (from reference): B=8, S=128, H=128, D_in=256
// ---------------------------------------------------------------------------
#define BB   8
#define SS   128
#define HH   128
#define DIN  256
#define MM   (BB*SS)          // 1024 global rows

// ---------------------------------------------------------------------------
// fp32 -> fp16 elementwise convert
// ---------------------------------------------------------------------------
__global__ void cvt_f32_f16_k(const float* __restrict__ src, f16* __restrict__ dst, int n) {
    int i = blockIdx.x * blockDim.x + threadIdx.x;
    if (i < n) dst[i] = (f16)src[i];
}

// Zero-padded f16 copy of W2 [4,128] -> [16,128] (rows 4..15 = 0)
__global__ void pad_w2_k(const float* __restrict__ W2, f16* __restrict__ w2p) {
    int i = blockIdx.x * blockDim.x + threadIdx.x;
    if (i < 16 * HH) w2p[i] = (i < 4 * HH) ? (f16)W2[i] : (f16)0;
}

// ---------------------------------------------------------------------------
// Fragment loader: 16-bit A/B layout for V_WMMA_F32_16X16X32_F16.
// lane 0-15 : row = lane,     holds K = k0+{0..7}  and k0+{16..23}
// lane16-31 : row = lane-16,  holds K = k0+{8..15} and k0+{24..31}
// Row-major source [*, ld], 16-byte aligned -> two b128 loads per lane.
// ---------------------------------------------------------------------------
__device__ __forceinline__ v16h load_frag(const f16* __restrict__ base, int row, int ld, int k) {
    const f16* p = base + row * ld + k;          // 32-bit index math
    v8h lo = *(const v8h*)(p);
    v8h hi = *(const v8h*)(p + 16);
    v16h r;
#pragma unroll
    for (int i = 0; i < 8; ++i) { r[i] = lo[i]; r[i + 8] = hi[i]; }
    return r;
}

__device__ __forceinline__ v8f wmma16(v16h a, v16h b, v8f c) {
    return __builtin_amdgcn_wmma_f32_16x16x32_f16(
        /*neg_a=*/false, a, /*neg_b=*/false, b,
        /*c_mod=*/(short)0, c, /*reuse_a=*/false, /*reuse_b=*/false);
}

// ---------------------------------------------------------------------------
// WMMA GEMM, 32x32 macro-tile per wave (2x2 of 16x16 => each fragment used
// twice: 8 b128 loads per 4 WMMAs).  C = scale*A(MxK)*W(NxK)^T + bias.
// Requires M%32==0, N%32==0, K%32==0.  Optional relu / f16 / f32 / transposed
// stores; outT layout: [row/128][col][row%128] (per-batch transposed).
// ---------------------------------------------------------------------------
__global__ void __launch_bounds__(256)
wmma_gemm32_k(const f16* __restrict__ A, const f16* __restrict__ W,
              const float* __restrict__ bias,
              f16* __restrict__ outH, float* __restrict__ outF, f16* __restrict__ outT,
              int tilesM2, int tilesN2, int K, int ldC,
              float scale, int relu,
              long strideA, long strideW, long strideC, int nBatch)
{
    int gid  = blockIdx.x * blockDim.x + threadIdx.x;
    int wave = gid >> 5;                          // wave32
    int lane = threadIdx.x & 31;
    int tpb  = tilesM2 * tilesN2;
    if (wave >= nBatch * tpb) return;

    int b    = wave / tpb;
    int rem  = wave - b * tpb;
    int tm   = rem / tilesN2;
    int tn   = rem - tm * tilesN2;

    const f16* Ab = A + (long)b * strideA;
    const f16* Wb = W + (long)b * strideW;

    int kb  = (lane < 16) ? 0 : 8;                // K sub-group base per layout
    int m0  = tm * 32 + (lane & 15), m1 = m0 + 16;
    int n0  = tn * 32 + (lane & 15), n1 = n0 + 16;

    v8f c00 = {}, c01 = {}, c10 = {}, c11 = {};
    for (int k0 = 0; k0 < K; k0 += 32) {
        v16h a0 = load_frag(Ab, m0, K, k0 + kb);
        v16h a1 = load_frag(Ab, m1, K, k0 + kb);
        v16h b0 = load_frag(Wb, n0, K, k0 + kb);
        v16h b1 = load_frag(Wb, n1, K, k0 + kb);
        c00 = wmma16(a0, b0, c00);
        c01 = wmma16(a0, b1, c01);
        c10 = wmma16(a1, b0, c10);
        c11 = wmma16(a1, b1, c11);
    }

    int half8   = (lane >> 4) * 8;                // C/D: lanes 16-31 hold rows +8
    int rowB0   = tm * 32 + half8;                // rows for c00/c01
    int rowB1   = rowB0 + 16;                     // rows for c10/c11
    int col0    = tn * 32 + (lane & 15);
    int col1    = col0 + 16;
    float bv0   = bias ? bias[col0] : 0.0f;
    float bv1   = bias ? bias[col1] : 0.0f;

#pragma unroll
    for (int t = 0; t < 4; ++t) {
        v8f  c    = (t == 0) ? c00 : (t == 1) ? c01 : (t == 2) ? c10 : c11;
        int  rb   = (t < 2) ? rowB0 : rowB1;
        int  col  = (t & 1) ? col1 : col0;
        float bv  = (t & 1) ? bv1 : bv0;
#pragma unroll
        for (int r = 0; r < 8; ++r) {
            int row = rb + r;
            float v = c[r] * scale + bv;
            if (relu) v = fmaxf(v, 0.0f);
            long ci = (long)b * strideC + row * ldC + col;
            if (outF) outF[ci] = v;
            if (outH) outH[ci] = (f16)v;
            if (outT) {                           // [row/128][col][row%128]
                int b2 = row >> 7, s = row & 127;
                outT[((long)b2 * HH + col) * SS + s] = (f16)v;
            }
        }
    }
}

// ---------------------------------------------------------------------------
// Final thin GEMM: out[1024,4] = h[1024,128] @ W2p[16,128]^T + b2.
// W2p is zero-padded to 16 rows, so all loads are unconditional; only the
// epilogue masks columns >= 4.  One 16x16 tile per wave, 64 waves.
// ---------------------------------------------------------------------------
__global__ void __launch_bounds__(256)
wmma_gemm_n4_k(const f16* __restrict__ A, const f16* __restrict__ W2p,
               const float* __restrict__ bias, float* __restrict__ outF)
{
    int gid  = blockIdx.x * blockDim.x + threadIdx.x;
    int wave = gid >> 5;
    int lane = threadIdx.x & 31;
    if (wave >= MM / 16) return;

    int kb = (lane < 16) ? 0 : 8;
    int m  = wave * 16 + (lane & 15);
    int n  = lane & 15;

    v8f c = {};
#pragma unroll
    for (int k0 = 0; k0 < HH; k0 += 32) {
        v16h a = load_frag(A,   m, HH, k0 + kb);
        v16h b = load_frag(W2p, n, HH, k0 + kb);
        c = wmma16(a, b, c);
    }

    if (n < 4) {
        float bv = bias[n];
        int half8 = (lane >> 4) * 8;
#pragma unroll
        for (int r = 0; r < 8; ++r) {
            int row = wave * 16 + half8 + r;
            outF[row * 4 + n] = c[r] + bv;
        }
    }
}

// ---------------------------------------------------------------------------
// Row softmax: one wave32 per row of 128 floats, in place.
// ---------------------------------------------------------------------------
__global__ void __launch_bounds__(256)
softmax_rows_k(float* __restrict__ attn, int rows)
{
    int gid  = blockIdx.x * blockDim.x + threadIdx.x;
    int wid  = gid >> 5;
    int lane = threadIdx.x & 31;
    if (wid >= rows) return;
    float* row = attn + wid * SS;

    float v[4];
    float m = -3.4e38f;
#pragma unroll
    for (int t = 0; t < 4; ++t) { v[t] = row[lane + 32 * t]; m = fmaxf(m, v[t]); }
#pragma unroll
    for (int o = 16; o > 0; o >>= 1) m = fmaxf(m, __shfl_xor(m, o, 32));
    float s = 0.0f;
#pragma unroll
    for (int t = 0; t < 4; ++t) { v[t] = __expf(v[t] - m); s += v[t]; }
#pragma unroll
    for (int o = 16; o > 0; o >>= 1) s += __shfl_xor(s, o, 32);
    float inv = 1.0f / s;
#pragma unroll
    for (int t = 0; t < 4; ++t) row[lane + 32 * t] = v[t] * inv;
}

// ---------------------------------------------------------------------------
// Fold conv1(1x3)+conv2(3x1) over the channel-broadcast input into per-output-
// channel 3x3 stencils Keff[o][dy*3+dx] and edge-dependent biases.
// bias3: [0][o]=interior, [1][o]=top row (i==0), [2][o]=bottom row (i==S-1).
// ---------------------------------------------------------------------------
__global__ void __launch_bounds__(128)
prep_conv_k(const float* __restrict__ c1w, const float* __restrict__ c1b,
            const float* __restrict__ c2w, const float* __restrict__ c2b,
            float* __restrict__ Keff, float* __restrict__ bias3)
{
    __shared__ float w1r[HH][3];
    __shared__ float c1bs[HH];
    int o = threadIdx.x;

    float s0 = 0.f, s1 = 0.f, s2 = 0.f;
    for (int cc = 0; cc < HH; ++cc) {             // conv1_w: [O,C,1,3]
        const float* p = c1w + (o * HH + cc) * 3;
        s0 += p[0]; s1 += p[1]; s2 += p[2];
    }
    w1r[o][0] = s0; w1r[o][1] = s1; w1r[o][2] = s2;
    c1bs[o] = c1b[o];
    __syncthreads();

    float K9[9]; float B2[3];
#pragma unroll
    for (int t = 0; t < 9; ++t) K9[t] = 0.f;
    B2[0] = B2[1] = B2[2] = 0.f;
    for (int cc = 0; cc < HH; ++cc) {             // conv2_w: [O,C,3,1]
        const float* q = c2w + (o * HH + cc) * 3;
        float r0 = w1r[cc][0], r1 = w1r[cc][1], r2 = w1r[cc][2], cb = c1bs[cc];
#pragma unroll
        for (int dy = 0; dy < 3; ++dy) {
            float w = q[dy];
            K9[dy * 3 + 0] += w * r0;
            K9[dy * 3 + 1] += w * r1;
            K9[dy * 3 + 2] += w * r2;
            B2[dy] += w * cb;
        }
    }
#pragma unroll
    for (int t = 0; t < 9; ++t) Keff[o * 9 + t] = K9[t];
    float cb2 = c2b[o];
    bias3[0 * HH + o] = cb2 + B2[0] + B2[1] + B2[2];   // interior rows
    bias3[1 * HH + o] = cb2 +        B2[1] + B2[2];    // i == 0   (dy=-1 pad)
    bias3[2 * HH + o] = cb2 + B2[0] + B2[1];           // i == S-1 (dy=+1 pad)
}

// ---------------------------------------------------------------------------
// interaction[b,i,j] = mean_o f( Keff[o] (*) attn_pad[b,i,j] + bias(o,i) )
// f(y): PReLU then (sigmoid(v)>0.6)*v  ==  v > ln(1.5) ? v : 0
// One thread per (b,i,j); Keff/bias3 staged in LDS (broadcast reads).
// ---------------------------------------------------------------------------
__global__ void __launch_bounds__(256)
stencil_mean_k(const float* __restrict__ attn, const float* __restrict__ Keff,
               const float* __restrict__ bias3, const float* __restrict__ prelu_a,
               f16* __restrict__ inter)
{
    __shared__ float Ks[HH * 9];
    __shared__ float Bs[3 * HH];
    int tid = threadIdx.x;
    for (int t = tid; t < HH * 9; t += 256) Ks[t] = Keff[t];
    for (int t = tid; t < 3 * HH; t += 256) Bs[t] = bias3[t];
    __syncthreads();

    float pa = prelu_a[0];
    int id = blockIdx.x * 256 + tid;
    int j = id & (SS - 1);
    int i = (id >> 7) & (SS - 1);
    int b = id >> 14;
    const float* ab = attn + b * SS * SS;

    float a[9];
#pragma unroll
    for (int dy = 0; dy < 3; ++dy)
#pragma unroll
        for (int dx = 0; dx < 3; ++dx) {
            int ii = i + dy - 1, jj = j + dx - 1;
            a[dy * 3 + dx] = (ii >= 0 && ii < SS && jj >= 0 && jj < SS)
                           ? ab[ii * SS + jj] : 0.0f;
        }

    const float* bsel = Bs + ((i == 0) ? HH : (i == SS - 1) ? 2 * HH : 0);
    const float T = 0.4054651081081644f;          // ln(0.6/0.4)
    float acc = 0.0f;
    for (int o = 0; o < HH; ++o) {
        const float* kp = Ks + o * 9;
        float y = bsel[o];
#pragma unroll
        for (int t = 0; t < 9; ++t) y += kp[t] * a[t];
        float v = (y >= 0.0f) ? y : pa * y;       // PReLU
        acc += (v > T) ? v : 0.0f;                // sigmoid mask * v
    }
    inter[id] = (f16)(acc * (1.0f / (float)HH));
}

// ---------------------------------------------------------------------------
// Host-side orchestration
// ---------------------------------------------------------------------------
extern "C" void kernel_launch(void* const* d_in, const int* in_sizes, int n_in,
                              void* d_out, int out_size, void* d_ws, size_t ws_size,
                              hipStream_t stream)
{
    const float* X      = (const float*)d_in[0];   // [8,128,256]
    const float* W_emb  = (const float*)d_in[1];   // [128,256]
    const float* b_emb  = (const float*)d_in[2];
    const float* W_q    = (const float*)d_in[3];   // [128,128]
    const float* b_q    = (const float*)d_in[4];
    const float* W_k    = (const float*)d_in[5];
    const float* b_k    = (const float*)d_in[6];
    const float* c1w    = (const float*)d_in[7];   // [128,128,1,3]
    const float* c1b    = (const float*)d_in[8];
    const float* c2w    = (const float*)d_in[9];   // [128,128,3,1]
    const float* c2b    = (const float*)d_in[10];
    const float* pre_a  = (const float*)d_in[11];  // scalar
    const float* W1     = (const float*)d_in[12];  // [128,128]
    const float* b1     = (const float*)d_in[13];
    const float* W2     = (const float*)d_in[14];  // [4,128]
    const float* b2     = (const float*)d_in[15];
    float* out          = (float*)d_out;           // [8,128,4]

    // ---- workspace layout (256B aligned) ----
    char* ws = (char*)d_ws;
    size_t off = 0;
    auto take = [&](size_t bytes) { char* p = ws + off; off = (off + bytes + 255) & ~(size_t)255; return p; };
    f16*   x_h     = (f16*)  take((size_t)MM * DIN * 2);
    f16*   emb_h   = (f16*)  take((size_t)MM * HH * 2);
    f16*   embT_h  = (f16*)  take((size_t)BB * HH * SS * 2);
    f16*   q_h     = (f16*)  take((size_t)MM * HH * 2);
    f16*   k_h     = (f16*)  take((size_t)MM * HH * 2);
    float* attn_f  = (float*)take((size_t)BB * SS * SS * 4);
    f16*   inter_h = (f16*)  take((size_t)BB * SS * SS * 2);
    f16*   fused_h = (f16*)  take((size_t)MM * HH * 2);
    f16*   hid_h   = (f16*)  take((size_t)MM * HH * 2);
    f16*   wemb_h  = (f16*)  take((size_t)HH * DIN * 2);
    f16*   wq_h    = (f16*)  take((size_t)HH * HH * 2);
    f16*   wk_h    = (f16*)  take((size_t)HH * HH * 2);
    f16*   w1_h    = (f16*)  take((size_t)HH * HH * 2);
    f16*   w2p_h   = (f16*)  take((size_t)16 * HH * 2);   // zero-padded W2
    float* Keff    = (float*)take((size_t)HH * 9 * 4);
    float* bias3   = (float*)take((size_t)3 * HH * 4);
    (void)ws_size; (void)in_sizes; (void)n_in; (void)out_size;

    auto cvt = [&](const float* s, f16* d, int n) {
        cvt_f32_f16_k<<<(n + 255) / 256, 256, 0, stream>>>(s, d, n);
    };
    auto gemm32 = [&](const f16* A, const f16* W, const float* bias,
                      f16* oH, float* oF, f16* oT,
                      int M, int K, int N, int ldC, float scale, int relu,
                      long sA, long sW, long sC, int nBatch) {
        int tilesM2 = M / 32, tilesN2 = N / 32;
        int waves   = nBatch * tilesM2 * tilesN2;
        int blocks  = (waves * 32 + 255) / 256;
        wmma_gemm32_k<<<blocks, 256, 0, stream>>>(A, W, bias, oH, oF, oT,
                tilesM2, tilesN2, K, ldC, scale, relu, sA, sW, sC, nBatch);
    };

    // 0) weight + input conversions to f16 (+ zero-padded W2)
    cvt(X,     x_h,    MM * DIN);
    cvt(W_emb, wemb_h, HH * DIN);
    cvt(W_q,   wq_h,   HH * HH);
    cvt(W_k,   wk_h,   HH * HH);
    cvt(W1,    w1_h,   HH * HH);
    pad_w2_k<<<(16 * HH + 255) / 256, 256, 0, stream>>>(W2, w2p_h);

    // 1) fold the two convs into per-channel 3x3 stencils
    prep_conv_k<<<1, 128, 0, stream>>>(c1w, c1b, c2w, c2b, Keff, bias3);

    // 2) embedded = X @ W_emb^T + b  (store f16 + transposed-per-batch f16)
    gemm32(x_h, wemb_h, b_emb, emb_h, nullptr, embT_h, MM, DIN, HH, HH, 1.0f, 0, 0, 0, 0, 1);

    // 3) query / key
    gemm32(emb_h, wq_h, b_q, q_h, nullptr, nullptr, MM, HH, HH, HH, 1.0f, 0, 0, 0, 0, 1);
    gemm32(emb_h, wk_h, b_k, k_h, nullptr, nullptr, MM, HH, HH, HH, 1.0f, 0, 0, 0, 0, 1);

    // 4) attn = softmax( q @ k^T / sqrt(H) )   (batched over B)
    const float scl = 0.08838834764831845f;       // 1/sqrt(128)
    gemm32(q_h, k_h, nullptr, nullptr, attn_f, nullptr, SS, HH, SS, SS, scl, 0,
           (long)SS * HH, (long)SS * HH, (long)SS * SS, BB);
    softmax_rows_k<<<(MM * 32 + 255) / 256, 256, 0, stream>>>(attn_f, MM);

    // 5) conv1+conv2+PReLU+mask, mean over channels -> interaction (f16)
    stencil_mean_k<<<(BB * SS * SS) / 256, 256, 0, stream>>>(attn_f, Keff, bias3, pre_a, inter_h);

    // 6) fused = interaction @ embedded   (B = embT rows => contiguous)
    gemm32(inter_h, embT_h, nullptr, fused_h, nullptr, nullptr, SS, SS, HH, HH, 1.0f, 0,
           (long)SS * SS, (long)HH * SS, (long)SS * HH, BB);

    // 7) h = relu(fused @ W1^T + b1)
    gemm32(fused_h, w1_h, b1, hid_h, nullptr, nullptr, MM, HH, HH, HH, 1.0f, 1, 0, 0, 0, 1);

    // 8) out = h @ W2p^T + b2   (N=4 via zero-padded 16-row W2)
    {
        int waves  = MM / 16;                     // 64 waves
        int blocks = (waves * 32 + 255) / 256;
        wmma_gemm_n4_k<<<blocks, 256, 0, stream>>>(hid_h, w2p_h, b2, out);
    }
}
